// GraphTransformerEdgeLayer_34351148433892
// MI455X (gfx1250) — matrix operand order repaired and checked by reading, not compiled, not run on hardware
//
#include <hip/hip_runtime.h>
#include <cstddef>
#include <cstdint>

#define NNODES 50000
#define NEDGES 500000
#define DMODEL 128

typedef __bf16 bf16;
typedef bf16  v16bf __attribute__((ext_vector_type(16)));
typedef float v8f   __attribute__((ext_vector_type(8)));
typedef int   v4i   __attribute__((vector_size(4 * sizeof(int))));

#define AS1 __attribute__((address_space(1)))
#define AS3 __attribute__((address_space(3)))

#if __has_builtin(__builtin_amdgcn_global_load_async_to_lds_b128)
#define HAVE_ASYNC_LDS 1
#endif

__device__ __forceinline__ bf16 tobf(float x) { return (bf16)x; }

// ---------------------------------------------------------------------------
// Pack fp32 weight W[K][NC] (row-major) into bf16 Wt[NC][K] (column-major).
// Lane fragment for WMMA B is then 16 contiguous bf16 (32B) per lane.
// ---------------------------------------------------------------------------
__global__ void pack_weight_kernel(const float* __restrict__ W, bf16* __restrict__ Wt,
                                   int K, int NC)
{
    int idx = blockIdx.x * blockDim.x + threadIdx.x;
    if (idx < K * NC) {
        int k = idx / NC;
        int n = idx - k * NC;
        Wt[(size_t)n * K + k] = tobf(W[idx]);
    }
}

__global__ void zero_kernel(float* __restrict__ p, size_t n)
{
    size_t stride = (size_t)gridDim.x * blockDim.x;
    for (size_t i = (size_t)blockIdx.x * blockDim.x + threadIdx.x; i < n; i += stride)
        p[i] = 0.0f;
}

// ---------------------------------------------------------------------------
// GEMM: C[M][NC] = act(A[M][KD] @ W[KD][NC] + bias) (+ resid)
// A fp32 row-major (converted to bf16 in-register), Wt bf16 column-major.
// Block = 128 threads = 4 waves; wave -> 16 rows x (NT*16) cols.
// The block's (NT*16)-column weight slab is staged in LDS via ASYNCcnt-tracked
// GLOBAL_LOAD_ASYNC_TO_LDS_B128, with a +8-element column pad so the 16 lanes
// of a B-fragment read hit 16 distinct LDS banks (272B stride).
// NT=8 for KD=128 (block covers 128 cols -> A read once for NC=128);
// NT=4 for KD=256 keeps the slab at ~34KB.
// ---------------------------------------------------------------------------
template <int KD, int NC, int NT, bool RELU>
__global__ __launch_bounds__(128)
void gemm_bf16_wmma(const float* __restrict__ A, const bf16* __restrict__ Wt,
                    const float* __restrict__ bias, const float* __restrict__ resid,
                    float* __restrict__ C, int M)
{
    constexpr int KDP  = KD + 8;             // padded column stride (272B/528B)
    constexpr int COLS = NT * 16;            // columns per block
    __shared__ bf16 sW[COLS * KDP];

    const int tid  = threadIdx.x;
    const int lane = tid & 31;
    const int wave = tid >> 5;
    const int half = lane >> 4;              // which 16-lane half of the wave
    const int l16  = lane & 15;

    const int mBase = blockIdx.x * 64 + wave * 16;
    const int nBase = blockIdx.y * COLS;

    // ---- stage COLS x KD bf16 weight slab into LDS (async) ----
    const bf16* __restrict__ slab = Wt + (size_t)nBase * KD;   // contiguous COLS*KD
    constexpr int CPC    = KD / 8;           // 16B chunks per column
    constexpr int CHUNKS = COLS * CPC;
    for (int i = tid; i < CHUNKS; i += 128) {
        const int col = i / CPC;
        const int w   = i - col * CPC;
#ifdef HAVE_ASYNC_LDS
        __builtin_amdgcn_global_load_async_to_lds_b128(
            (AS1 v4i*)(slab + (size_t)col * KD + w * 8),
            (AS3 v4i*)(&sW[col * KDP + w * 8]), 0, 0);
#else
        *(float4*)(&sW[col * KDP + w * 8]) =
            *(const float4*)(slab + (size_t)col * KD + w * 8);
#endif
    }
#ifdef HAVE_ASYNC_LDS
#if __has_builtin(__builtin_amdgcn_s_wait_asynccnt)
    __builtin_amdgcn_s_wait_asynccnt(0);
#else
    asm volatile("s_wait_asynccnt 0" ::: "memory");
#endif
#endif
    __syncthreads();

    int arow = mBase + l16;
    if (arow > M - 1) arow = M - 1;          // clamp reads; stores guarded below
    const float* __restrict__ Arow = A + (size_t)arow * KD;

    v8f acc[NT];
#pragma unroll
    for (int nt = 0; nt < NT; ++nt)
#pragma unroll
        for (int i = 0; i < 8; ++i) acc[nt][i] = 0.0f;

#pragma unroll
    for (int k0 = 0; k0 < KD; k0 += 32) {
        // A fragment (contiguous k order, matching B):
        //  elems 0..15 -> K = k0 + half*16 + t
        const float4 a0 = *(const float4*)(Arow + k0 + half * 16);
        const float4 a1 = *(const float4*)(Arow + k0 + half * 16 + 4);
        const float4 a2 = *(const float4*)(Arow + k0 + half * 16 + 8);
        const float4 a3 = *(const float4*)(Arow + k0 + half * 16 + 12);
        v16bf av;
        av[0]  = tobf(a0.x); av[1]  = tobf(a0.y); av[2]  = tobf(a0.z); av[3]  = tobf(a0.w);
        av[4]  = tobf(a1.x); av[5]  = tobf(a1.y); av[6]  = tobf(a1.z); av[7]  = tobf(a1.w);
        av[8]  = tobf(a2.x); av[9]  = tobf(a2.y); av[10] = tobf(a2.z); av[11] = tobf(a2.w);
        av[12] = tobf(a3.x); av[13] = tobf(a3.y); av[14] = tobf(a3.z); av[15] = tobf(a3.w);

        // Batch all B-fragment LDS loads, then chain the WMMAs.
        v16bf bv[NT];
#pragma unroll
        for (int nt = 0; nt < NT; ++nt)
            bv[nt] = *(const v16bf*)(&sW[(nt * 16 + l16) * KDP + k0 + half * 16]);
#pragma unroll
        for (int nt = 0; nt < NT; ++nt)
            acc[nt] = __builtin_amdgcn_wmma_f32_16x16x32_bf16(
                false, av, false, bv[nt], (short)0, acc[nt], false, false);
    }

    // Epilogue: C/D layout -> VGPR r holds row (r + 8*half), col l16
#pragma unroll
    for (int nt = 0; nt < NT; ++nt) {
        const int n = nBase + nt * 16 + l16;
#pragma unroll
        for (int r = 0; r < 8; ++r) {
            const int row = mBase + r + half * 8;
            if (row < M) {
                float x = acc[nt][r];
                if (bias) x += bias[n];
                if (RELU) x = x > 0.0f ? x : 0.0f;
                if (resid) x += resid[(size_t)row * NC + n];
                C[(size_t)row * NC + n] = x;
            }
        }
    }
}

// ---------------------------------------------------------------------------
// Edge attention: score = K[src]*Q[dst]/sqrt(16)*pe ; e_attn = score (in place
// over pe); s = exp(clip(sum_head score)); msg = V[src]*env*s; segment-sum
// into wV[dst], z[dst] via fp32 atomics. Per-head sum = 16-lane shfl reduce.
// ---------------------------------------------------------------------------
__global__ __launch_bounds__(256)
void edge_attn_kernel(const float* __restrict__ Qm, const float* __restrict__ Km,
                      const float* __restrict__ Vm, float* __restrict__ pe,
                      const float* __restrict__ env, const int* __restrict__ src,
                      const int* __restrict__ dst, float* __restrict__ wV,
                      float* __restrict__ z, int E)
{
    const float rsdh = 0.25f; // 1/sqrt(16)
    const size_t total = (size_t)E * DMODEL;
    const size_t stride = (size_t)gridDim.x * blockDim.x;
    for (size_t i = (size_t)blockIdx.x * blockDim.x + threadIdx.x; i < total; i += stride) {
        const int eidx = (int)(i >> 7);
        const int c    = (int)(i & 127);
        const int s    = src[eidx];
        const int d    = dst[eidx];
        const float score = Km[(size_t)s * DMODEL + c] * Qm[(size_t)d * DMODEL + c]
                            * rsdh * pe[i];
        pe[i] = score;                       // e_attn (in place)
        float hs = score;                    // sum over the 16 channels of a head
        hs += __shfl_xor(hs, 1, 16);
        hs += __shfl_xor(hs, 2, 16);
        hs += __shfl_xor(hs, 4, 16);
        hs += __shfl_xor(hs, 8, 16);
        hs = fminf(5.0f, fmaxf(-5.0f, hs));
        const float sexp = __expf(hs);
        const float msg  = Vm[(size_t)s * DMODEL + c] * env[eidx] * sexp;
        atomicAdd(&wV[(size_t)d * DMODEL + c], msg);
        if ((c & 15) == 0)
            atomicAdd(&z[(size_t)d * 8 + (c >> 4)], sexp);
    }
}

__global__ void vattn_kernel(const float* __restrict__ wV, const float* __restrict__ z,
                             float* __restrict__ vat, int N)
{
    const size_t total = (size_t)N * DMODEL;
    const size_t stride = (size_t)gridDim.x * blockDim.x;
    for (size_t i = (size_t)blockIdx.x * blockDim.x + threadIdx.x; i < total; i += stride) {
        const size_t n = i >> 7;
        const int c = (int)(i & 127);
        vat[i] = wV[i] / (z[n * 8 + (c >> 4)] + 1e-6f);
    }
}

// ---------------------------------------------------------------------------
// BatchNorm (training-mode, biased batch stats over rows)
// ---------------------------------------------------------------------------
__global__ __launch_bounds__(128)
void bn_stats_kernel(const float* __restrict__ X, float* __restrict__ stats, int M)
{
    const int c = threadIdx.x;                 // one thread per column
    float s = 0.0f, s2 = 0.0f;
    for (int r = blockIdx.x; r < M; r += gridDim.x) {
        const float x = X[(size_t)r * DMODEL + c];
        s += x; s2 += x * x;
    }
    atomicAdd(&stats[c], s);
    atomicAdd(&stats[DMODEL + c], s2);
}

__global__ __launch_bounds__(256)
void bn_apply_kernel(const float* __restrict__ X, const float* __restrict__ stats,
                     const float* __restrict__ g, const float* __restrict__ b,
                     float* __restrict__ Y, int M, float invM)
{
    const size_t total = (size_t)M * DMODEL;
    const size_t stride = (size_t)gridDim.x * blockDim.x;
    for (size_t i = (size_t)blockIdx.x * blockDim.x + threadIdx.x; i < total; i += stride) {
        const int c = (int)(i & 127);
        const float m   = stats[c] * invM;
        const float var = stats[DMODEL + c] * invM - m * m;
        Y[i] = (X[i] - m) * rsqrtf(var + 1e-5f) * g[c] + b[c];
    }
}

// ---------------------------------------------------------------------------
static void run_bn(const float* X, const float* g, const float* b, float* Y,
                   int M, float* stats, hipStream_t stream)
{
    zero_kernel<<<1, 256, 0, stream>>>(stats, 2 * DMODEL);
    bn_stats_kernel<<<512, DMODEL, 0, stream>>>(X, stats, M);
    bn_apply_kernel<<<4096, 256, 0, stream>>>(X, stats, g, b, Y, M, 1.0f / (float)M);
}

extern "C" void kernel_launch(void* const* d_in, const int* in_sizes, int n_in,
                              void* d_out, int out_size, void* d_ws, size_t ws_size,
                              hipStream_t stream)
{
    const float* v    = (const float*)d_in[0];
    const float* e    = (const float*)d_in[1];
    const float* env  = (const float*)d_in[2];
    const int*   src  = (const int*)d_in[3];
    const int*   dst  = (const int*)d_in[4];
    const float* WQ   = (const float*)d_in[5];
    const float* WK   = (const float*)d_in[6];
    const float* WV   = (const float*)d_in[7];
    const float* We   = (const float*)d_in[8];
    const float* WOv  = (const float*)d_in[9];
    const float* bOv  = (const float*)d_in[10];
    const float* WOe  = (const float*)d_in[11];
    const float* bOe  = (const float*)d_in[12];
    const float* g1v  = (const float*)d_in[13];
    const float* bn1v = (const float*)d_in[14];
    const float* g1e  = (const float*)d_in[15];
    const float* bn1e = (const float*)d_in[16];
    const float* g2v  = (const float*)d_in[17];
    const float* bn2v = (const float*)d_in[18];
    const float* g2e  = (const float*)d_in[19];
    const float* bn2e = (const float*)d_in[20];
    const float* W1v  = (const float*)d_in[21];
    const float* fb1v = (const float*)d_in[22];
    const float* W2v  = (const float*)d_in[23];
    const float* fb2v = (const float*)d_in[24];
    const float* W1e  = (const float*)d_in[25];
    const float* fb1e = (const float*)d_in[26];
    const float* W2e  = (const float*)d_in[27];
    const float* fb2e = (const float*)d_in[28];

    float* out_v = (float*)d_out;
    float* out_e = out_v + (size_t)NNODES * DMODEL;

    // ---- workspace carve ----
    char* base = (char*)d_ws;
    size_t off = 0;
#define CARVE(var, type, count)                                            \
    type* var = (type*)(base + off);                                       \
    off = (off + (size_t)(count) * sizeof(type) + 255) & ~(size_t)255;

    CARVE(Qb,  float, (size_t)NNODES * DMODEL)     // Q ; reused as t2_v
    CARVE(Kb,  float, (size_t)NNODES * DMODEL)
    CARVE(Vb,  float, (size_t)NNODES * DMODEL)
    CARVE(wV,  float, (size_t)NNODES * DMODEL)
    CARVE(zb,  float, (size_t)NNODES * 8)
    CARVE(vat, float, (size_t)NNODES * DMODEL)
    CARVE(tv,  float, (size_t)NNODES * DMODEL)     // t_v ; v1 in place
    CARVE(hv,  float, (size_t)NNODES * 2 * DMODEL)
    CARVE(pe,  float, (size_t)NEDGES * DMODEL)     // pe -> e_attn -> t2_e
    CARVE(te,  float, (size_t)NEDGES * DMODEL)     // t_e ; e1 in place
    CARVE(he,  float, (size_t)NEDGES * 2 * DMODEL)
    CARVE(stats, float, 2 * DMODEL)
    CARVE(wqT,  bf16, DMODEL * DMODEL)
    CARVE(wkT,  bf16, DMODEL * DMODEL)
    CARVE(wvT,  bf16, DMODEL * DMODEL)
    CARVE(weT,  bf16, DMODEL * DMODEL)
    CARVE(wovT, bf16, DMODEL * DMODEL)
    CARVE(woeT, bf16, DMODEL * DMODEL)
    CARVE(w1vT, bf16, DMODEL * 2 * DMODEL)
    CARVE(w2vT, bf16, DMODEL * 2 * DMODEL)
    CARVE(w1eT, bf16, DMODEL * 2 * DMODEL)
    CARVE(w2eT, bf16, DMODEL * 2 * DMODEL)
#undef CARVE
    (void)ws_size; (void)in_sizes; (void)n_in; (void)out_size;

    // ---- pack weights to bf16 column-major ----
    const int pk1 = (DMODEL * DMODEL + 255) / 256;
    const int pk2 = (DMODEL * 2 * DMODEL + 255) / 256;
    pack_weight_kernel<<<pk1, 256, 0, stream>>>(WQ,  wqT,  DMODEL, DMODEL);
    pack_weight_kernel<<<pk1, 256, 0, stream>>>(WK,  wkT,  DMODEL, DMODEL);
    pack_weight_kernel<<<pk1, 256, 0, stream>>>(WV,  wvT,  DMODEL, DMODEL);
    pack_weight_kernel<<<pk1, 256, 0, stream>>>(We,  weT,  DMODEL, DMODEL);
    pack_weight_kernel<<<pk1, 256, 0, stream>>>(WOv, wovT, DMODEL, DMODEL);
    pack_weight_kernel<<<pk1, 256, 0, stream>>>(WOe, woeT, DMODEL, DMODEL);
    pack_weight_kernel<<<pk2, 256, 0, stream>>>(W1v, w1vT, DMODEL, 2 * DMODEL);
    pack_weight_kernel<<<pk2, 256, 0, stream>>>(W2v, w2vT, 2 * DMODEL, DMODEL);
    pack_weight_kernel<<<pk2, 256, 0, stream>>>(W1e, w1eT, DMODEL, 2 * DMODEL);
    pack_weight_kernel<<<pk2, 256, 0, stream>>>(W2e, w2eT, 2 * DMODEL, DMODEL);

    const dim3 blk(128, 1, 1);
    const dim3 gN((NNODES + 63) / 64, 1);      // KD=128, NC=128, NT=8 -> y=1
    const dim3 gE((NEDGES + 63) / 64, 1);
    const dim3 gNh((NNODES + 63) / 64, 2);     // KD=128, NC=256, NT=8 -> y=2
    const dim3 gEh((NEDGES + 63) / 64, 2);
    const dim3 gN2((NNODES + 63) / 64, 2);     // KD=256, NC=128, NT=4 -> y=2
    const dim3 gE2((NEDGES + 63) / 64, 2);

    // ---- attention projections ----
    gemm_bf16_wmma<DMODEL, DMODEL, 8, false><<<gN, blk, 0, stream>>>(v, wqT, nullptr, nullptr, Qb, NNODES);
    gemm_bf16_wmma<DMODEL, DMODEL, 8, false><<<gN, blk, 0, stream>>>(v, wkT, nullptr, nullptr, Kb, NNODES);
    gemm_bf16_wmma<DMODEL, DMODEL, 8, false><<<gN, blk, 0, stream>>>(v, wvT, nullptr, nullptr, Vb, NNODES);
    gemm_bf16_wmma<DMODEL, DMODEL, 8, false><<<gE, blk, 0, stream>>>(e, weT, nullptr, nullptr, pe, NEDGES);

    // ---- edge attention + segment sums ----
    zero_kernel<<<4096, 256, 0, stream>>>(wV, (size_t)NNODES * DMODEL);
    zero_kernel<<<1024, 256, 0, stream>>>(zb, (size_t)NNODES * 8);
    edge_attn_kernel<<<4096, 256, 0, stream>>>(Qb, Kb, Vb, pe, env, src, dst, wV, zb, NEDGES);
    vattn_kernel<<<4096, 256, 0, stream>>>(wV, zb, vat, NNODES);

    // ---- O projection + residual + BN ----
    gemm_bf16_wmma<DMODEL, DMODEL, 8, false><<<gN, blk, 0, stream>>>(vat, wovT, bOv, v, tv, NNODES);
    run_bn(tv, g1v, bn1v, tv, NNODES, stats, stream);           // v1 in place
    gemm_bf16_wmma<DMODEL, DMODEL, 8, false><<<gE, blk, 0, stream>>>(pe, woeT, bOe, e, te, NEDGES);
    run_bn(te, g1e, bn1e, te, NEDGES, stats, stream);           // e1 in place

    // ---- FFN (v) + residual + BN ----
    gemm_bf16_wmma<DMODEL, 2 * DMODEL, 8, true><<<gNh, blk, 0, stream>>>(tv, w1vT, fb1v, nullptr, hv, NNODES);
    gemm_bf16_wmma<2 * DMODEL, DMODEL, 4, false><<<gN2, blk, 0, stream>>>(hv, w2vT, fb2v, tv, Qb, NNODES);
    run_bn(Qb, g2v, bn2v, out_v, NNODES, stats, stream);

    // ---- FFN (e) + residual + BN ----
    gemm_bf16_wmma<DMODEL, 2 * DMODEL, 8, true><<<gEh, blk, 0, stream>>>(te, w1eT, fb1e, nullptr, he, NEDGES);
    gemm_bf16_wmma<2 * DMODEL, DMODEL, 4, false><<<gE2, blk, 0, stream>>>(he, w2eT, fb2e, te, pe, NEDGES);
    run_bn(pe, g2e, bn2e, out_e, NEDGES, stats, stream);
}
